// SelfAttentionConcat_61297773248561
// MI455X (gfx1250) — matrix operand order repaired and checked by reading, not compiled, not run on hardware
//
#include <hip/hip_runtime.h>

typedef unsigned short u16;
typedef __bf16 bf16_t;
typedef bf16_t v16bf __attribute__((ext_vector_type(16)));
typedef float v8f __attribute__((ext_vector_type(8)));
typedef u16 u16x8 __attribute__((ext_vector_type(8)));

union Frag {
  v16bf v;
  u16x8 h[2];
};

__device__ __forceinline__ u16 f2bf(float f) {
  unsigned u = __float_as_uint(f);
  u += 0x7FFFu + ((u >> 16) & 1u);   // round-to-nearest-even
  return (u16)(u >> 16);
}

#define WMMA_BF16(A, B, Cacc) \
  __builtin_amdgcn_wmma_f32_16x16x32_bf16(false, (A), false, (B), (short)0, (Cacc), false, false)

static constexpr int Bn = 4, Cc = 64, C3 = 192, Hh = 64, Ww = 64, HW = 4096;
static constexpr int Hp = 66, Wp = 66;                               // zero-halo padded
static constexpr size_t WQKV_STRIDE = (size_t)9 * 6 * 12 * 32 * 16;  // 331776 halves per conv

// ---------------------------------------------------------------------------
// Kernel 1: concat fm1/fm2/fm3 (NCHW f32) -> fmT (padded NHWC bf16)
//   fmT[b][hp=h+1][wp=w+1][c], 1-pixel zero halo => conv loads never branch.
// ---------------------------------------------------------------------------
__global__ void prep_fm_kernel(const float* __restrict__ f1, const float* __restrict__ f2,
                               const float* __restrict__ f3, u16* __restrict__ fmT, int total) {
  int idx = blockIdx.x * blockDim.x + threadIdx.x;
  if (idx >= total) return;
  int c = idx % C3; int t = idx / C3;
  int wp = t % Wp; t /= Wp;
  int hp = t % Hp; t /= Hp;
  int b = t;
  int h = hp - 1, w = wp - 1;
  u16 v = 0;
  if (h >= 0 && h < Hh && w >= 0 && w < Ww) {
    const float* src = (c < 64) ? f1 : (c < 128 ? f2 : f3);
    int cin = c & 63;
    v = f2bf(src[(((size_t)b * Cc + cin) * Hh + h) * Ww + w]);
  }
  fmT[idx] = v;
}

// ---------------------------------------------------------------------------
// Kernel 2: pack wq/wk/wv (OIHW f32) into WMMA A-fragment order, bf16:
//   fragW[qkv][p=kh*3+kw][kb(6)][mb(12)][lane(32)][i(16)]
//   A layout: lane -> M = co = mb*16+(lane&15);  halves i -> K = ci pattern
// ---------------------------------------------------------------------------
__global__ void prep_w_kernel(const float* __restrict__ wq, const float* __restrict__ wk,
                              const float* __restrict__ wv, u16* __restrict__ fragW, int total) {
  int idx = blockIdx.x * blockDim.x + threadIdx.x;
  if (idx >= total) return;
  int i = idx % 16; int t = idx / 16;
  int lane = t % 32; t /= 32;
  int mb = t % 12; t /= 12;
  int kb = t % 6; t /= 6;
  int p = t % 9; t /= 9;
  int qkv = t;
  int grp = lane >> 4, ln = lane & 15;
  int co = mb * 16 + ln;
  int ci = kb * 32 + grp * 16 + i;   // contiguous K: k = 16*grp + i
  int kh = p / 3, kw = p % 3;
  const float* src = (qkv == 0) ? wq : (qkv == 1 ? wk : wv);
  fragW[idx] = f2bf(src[(size_t)(co * C3 + ci) * 9 + kh * 3 + kw]);
}

// ---------------------------------------------------------------------------
// Kernel 3: implicit-GEMM 3x3 conv producing Q,K,V in one pass.
// One wave = 16 output channels x 16 spatial positions (one row segment).
// All inner-loop addresses are (base VGPR64 + compile-time immediate):
//   image  : ibase + ((p/3)*Wp + p%3)*C3 + kb*32       (halo => unconditional)
//   weights: wbase + (p*6+kb)*6144 (+ qkv bank strides)
// Q,K stored as [b][n][c] bf16 ; V stored as [b][c][n] bf16.
// ---------------------------------------------------------------------------
__global__ __launch_bounds__(128) void conv_qkv_kernel(
    const u16* __restrict__ fmT, const u16* __restrict__ fragW,
    const float* __restrict__ bq, const float* __restrict__ bk, const float* __restrict__ bv,
    u16* __restrict__ Qt, u16* __restrict__ Kt, u16* __restrict__ Vc) {
  int lane = threadIdx.x & 31;
  int id = blockIdx.x * 4 + (threadIdx.x >> 5);   // 12288 waves
  int mb = id % 12; id /= 12;
  int wt = id % 4;  id /= 4;
  int h  = id % Hh; id /= Hh;
  int b  = id;
  int grp = lane >> 4, ln = lane & 15;
  int w0 = wt * 16;

  v8f aq, ak, av;   // D-layout: vgpr r -> co = mb*16 + grp*8 + r ; lane&15 -> spatial
  const float* bqb = bq + mb * 16 + grp * 8;
  const float* bkb = bk + mb * 16 + grp * 8;
  const float* bvb = bv + mb * 16 + grp * 8;
#pragma unroll
  for (int r = 0; r < 8; ++r) { aq[r] = bqb[r]; ak[r] = bkb[r]; av[r] = bvb[r]; }

  // base pointers: all runtime terms hoisted, inner offsets are immediates
  const u16* ibase = fmT + ((size_t)(b * Hp + h) * Wp + (w0 + ln)) * C3 + grp * 16;
  const u16* wbase = fragW + ((size_t)mb * 32 + lane) * 16;

#pragma unroll
  for (int p = 0; p < 9; ++p) {
#pragma unroll
    for (int kb = 0; kb < 6; ++kb) {
      Frag bi;   // B layout: lane -> N = spatial ; halves -> K = ci (16 contiguous)
      const u16x8* pp = (const u16x8*)(ibase + (size_t)((p / 3) * Wp + (p % 3)) * C3 + kb * 32);
      bi.h[0] = pp[0]; bi.h[1] = pp[1];
      const u16* wb = wbase + (size_t)(p * 6 + kb) * 6144;   // 12*32*16 halves per step
      Frag faq, fak, fav;
      faq.h[0] = ((const u16x8*)wb)[0];
      faq.h[1] = ((const u16x8*)wb)[1];
      fak.h[0] = ((const u16x8*)(wb + WQKV_STRIDE))[0];
      fak.h[1] = ((const u16x8*)(wb + WQKV_STRIDE))[1];
      fav.h[0] = ((const u16x8*)(wb + 2 * WQKV_STRIDE))[0];
      fav.h[1] = ((const u16x8*)(wb + 2 * WQKV_STRIDE))[1];
      aq = WMMA_BF16(faq.v, bi.v, aq);
      ak = WMMA_BF16(fak.v, bi.v, ak);
      av = WMMA_BF16(fav.v, bi.v, av);
    }
  }

  int n = h * Ww + w0 + ln;
  u16x8 pq, pk;
#pragma unroll
  for (int r = 0; r < 8; ++r) { pq[r] = f2bf(aq[r]); pk[r] = f2bf(ak[r]); }
  *(u16x8*)(Qt + ((size_t)b * HW + n) * C3 + mb * 16 + grp * 8) = pq;
  *(u16x8*)(Kt + ((size_t)b * HW + n) * C3 + mb * 16 + grp * 8) = pk;
  u16* vdst = Vc + ((size_t)b * C3 + mb * 16 + grp * 8) * HW + n;
#pragma unroll
  for (int r = 0; r < 8; ++r) vdst[(size_t)r * HW] = f2bf(av[r]);
}

// ---------------------------------------------------------------------------
// Kernel 4: flash attention over spatial positions. One wave = 16 query rows.
// scores = Qt * Kt^T (K-dim = 192 channels); online softmax; out += P * V.
// K/V pointers bumped additively per chunk; all tile offsets are immediates.
// Output written f32 in [b][n][c] flat order (== reference reshape).
// ---------------------------------------------------------------------------
__global__ __launch_bounds__(128) void attn_kernel(
    const u16* __restrict__ Qt, const u16* __restrict__ Kt,
    const u16* __restrict__ Vc, float* __restrict__ out) {
  __shared__ u16 Plds[4 * 16 * 64];
  int lane = threadIdx.x & 31, wid = threadIdx.x >> 5;
  int id = blockIdx.x * 4 + wid;                 // 1024 waves
  int nt = id % 256, b = id / 256;
  int nB = nt * 16;
  int grp = lane >> 4, ln = lane & 15;
  const u16* Qb = Qt + (size_t)b * HW * C3;
  u16* P = Plds + wid * 16 * 64;
  u16* Pst = P + (size_t)(grp * 8) * 64 + ln;        // softmax-weight store base
  const u16* Pld = P + (size_t)ln * 64 + grp * 16;   // A-frag load base

  // Q A-frags resident for the whole sweep (M = query row, K = channel)
  Frag fq[6];
  const u16* qb = Qb + (size_t)(nB + ln) * C3 + grp * 16;
#pragma unroll
  for (int kb = 0; kb < 6; ++kb) {
    const u16x8* p0 = (const u16x8*)(qb + kb * 32);
    fq[kb].h[0] = p0[0];
    fq[kb].h[1] = p0[1];   // +8 halves
  }

  v8f acc[12];
#pragma unroll
  for (int ct = 0; ct < 12; ++ct)
#pragma unroll
    for (int r = 0; r < 8; ++r) acc[ct][r] = 0.f;
  float runM[8], runS[8];
#pragma unroll
  for (int r = 0; r < 8; ++r) { runM[r] = -1e30f; runS[r] = 0.f; }

  const u16* kptr = Kt + (size_t)b * HW * C3 + (size_t)ln * C3 + grp * 16;
  const u16* vptr = Vc + (size_t)b * C3 * HW + (size_t)ln * HW + grp * 16;

  for (int mc = 0; mc < HW; mc += 64) {
    // --- scores: 4 tiles of 16 keys, K-dim = 192 ---
    v8f s[4];
#pragma unroll
    for (int t = 0; t < 4; ++t) {
      v8f sv;
#pragma unroll
      for (int r = 0; r < 8; ++r) sv[r] = 0.f;
#pragma unroll
      for (int kb = 0; kb < 6; ++kb) {
        Frag fk;   // B layout: lane -> key col ; halves -> channel (16 contiguous)
        const u16x8* pp = (const u16x8*)(kptr + (size_t)t * 16 * C3 + kb * 32);
        fk.h[0] = pp[0]; fk.h[1] = pp[1];
        sv = WMMA_BF16(fq[kb].v, fk.v, sv);
      }
      s[t] = sv;
    }
    // --- online softmax stats (row r lives in vgpr r, lane group = M-half) ---
    float nmax[8], scale[8];
#pragma unroll
    for (int r = 0; r < 8; ++r) {
      float m = fmaxf(fmaxf(s[0][r], s[1][r]), fmaxf(s[2][r], s[3][r]));
      m = fmaxf(m, __shfl_xor(m, 1, 32));
      m = fmaxf(m, __shfl_xor(m, 2, 32));
      m = fmaxf(m, __shfl_xor(m, 4, 32));
      m = fmaxf(m, __shfl_xor(m, 8, 32));
      float nm = fmaxf(runM[r], m);
      nmax[r] = nm;
      scale[r] = __expf(runM[r] - nm);
      runM[r] = nm;
    }
    float csum[8];
#pragma unroll
    for (int r = 0; r < 8; ++r) csum[r] = 0.f;
#pragma unroll
    for (int t = 0; t < 4; ++t)
#pragma unroll
      for (int r = 0; r < 8; ++r) {
        float pe = __expf(s[t][r] - nmax[r]);
        csum[r] += pe;
        Pst[r * 64 + t * 16] = f2bf(pe);   // transpose via LDS (imm offsets)
      }
#pragma unroll
    for (int r = 0; r < 8; ++r) {
      float cs = csum[r];
      cs += __shfl_xor(cs, 1, 32);
      cs += __shfl_xor(cs, 2, 32);
      cs += __shfl_xor(cs, 4, 32);
      cs += __shfl_xor(cs, 8, 32);
      runS[r] = runS[r] * scale[r] + cs;
#pragma unroll
      for (int ct = 0; ct < 12; ++ct) acc[ct][r] *= scale[r];
    }
    asm volatile("s_wait_dscnt 0" ::: "memory");
    // --- out += P * V  (K-dim = 64 keys in 2 chunks) ---
#pragma unroll
    for (int kc = 0; kc < 2; ++kc) {
      Frag fp;   // A layout: lane -> query row ; halves -> key index
      const u16x8* lp = (const u16x8*)(Pld + kc * 32);
      fp.h[0] = lp[0];
      fp.h[1] = lp[1];   // +8 halves
#pragma unroll
      for (int ct = 0; ct < 12; ++ct) {
        Frag fv;   // B layout: lane -> channel col ; halves -> key (16 contiguous)
        const u16x8* vp = (const u16x8*)(vptr + (size_t)ct * 16 * HW + kc * 32);
        fv.h[0] = vp[0]; fv.h[1] = vp[1];
        acc[ct] = WMMA_BF16(fp.v, fv.v, acc[ct]);
      }
    }
    kptr += 64 * C3;
    vptr += 64;
  }
  // --- epilogue: divide by running sum, write [b][n][c] f32 (imm offsets) ---
  float* obase = out + ((size_t)b * HW + nB + grp * 8) * C3 + ln;
#pragma unroll
  for (int r = 0; r < 8; ++r) {
    float inv = 1.0f / runS[r];
#pragma unroll
    for (int ct = 0; ct < 12; ++ct) obase[(size_t)r * C3 + ct * 16] = acc[ct][r] * inv;
  }
}

// ---------------------------------------------------------------------------
extern "C" void kernel_launch(void* const* d_in, const int* in_sizes, int n_in,
                              void* d_out, int out_size, void* d_ws, size_t ws_size,
                              hipStream_t stream) {
  const float* fm1 = (const float*)d_in[0];
  const float* fm2 = (const float*)d_in[1];
  const float* fm3 = (const float*)d_in[2];
  const float* wq = (const float*)d_in[3];
  const float* bq = (const float*)d_in[4];
  const float* wk = (const float*)d_in[5];
  const float* bk = (const float*)d_in[6];
  const float* wv = (const float*)d_in[7];
  const float* bv = (const float*)d_in[8];

  const size_t fmT_elems = (size_t)Bn * Hp * Wp * C3;       // 3,345,408 (padded)
  const size_t fragW_elems = 3 * WQKV_STRIDE;               //   995,328
  const size_t qkv_elems = (size_t)Bn * HW * C3;            // 3,145,728

  char* ws = (char*)d_ws;
  size_t off = 0;
  u16* fmT = (u16*)(ws + off);   off += fmT_elems * sizeof(u16);
  off = (off + 255) & ~(size_t)255;
  u16* fragW = (u16*)(ws + off); off += fragW_elems * sizeof(u16);
  off = (off + 255) & ~(size_t)255;
  u16* Qt = (u16*)(ws + off);    off += qkv_elems * sizeof(u16);
  u16* Kt = (u16*)(ws + off);    off += qkv_elems * sizeof(u16);
  u16* Vc = (u16*)(ws + off);    off += qkv_elems * sizeof(u16);

  prep_fm_kernel<<<(int)((fmT_elems + 255) / 256), 256, 0, stream>>>(fm1, fm2, fm3, fmT,
                                                                     (int)fmT_elems);
  prep_w_kernel<<<(int)((fragW_elems + 255) / 256), 256, 0, stream>>>(wq, wk, wv, fragW,
                                                                      (int)fragW_elems);
  conv_qkv_kernel<<<3072, 128, 0, stream>>>(fmT, fragW, bq, bk, bv, Qt, Kt, Vc);
  attn_kernel<<<256, 128, 0, stream>>>(Qt, Kt, Vc, (float*)d_out);
}
